// ProteinFeatures_67362267070497
// MI455X (gfx1250) — compile-verified, matrix operand order checked by reading, not run on hardware
//
#include <hip/hip_runtime.h>

typedef __bf16  v16bf __attribute__((ext_vector_type(16)));
typedef float   v8f   __attribute__((ext_vector_type(8)));

#define LRES   1024
#define TOPK   48
#define NEDGE  (LRES * TOPK)      // 49152
#define EF     128
#define KE     416                // 16 pos + 16 rbf(D) + 24*16 pair rbf
#define KEP    512                // KE zero-padded to 4x128 phases
#define KS     1280               // 5*32*8 (already divisible by 128)

// pair list (indices into coords5 order: N=0, Ca=1, C=2, O=3, Cb=4)
__constant__ int PAIR_A[24] = {0,2,3,4,1,1,1,1,0,0,0,4,4,3,0,2,3,4,2,3,4,2,3,2};
__constant__ int PAIR_B[24] = {0,2,3,4,0,2,3,4,2,3,4,2,3,2,1,1,1,1,0,0,0,4,4,3};

__device__ __forceinline__ unsigned short f32_to_bf16(float f) {
    unsigned u = __float_as_uint(f);
    unsigned rnd = 0x7fffu + ((u >> 16) & 1u);
    return (unsigned short)((u + rnd) >> 16);
}

// ---------------------------------------------------------------------------
// Kernel 1: per-residue derived coords  (N, Ca, C, O, Cb) -> coords5[1024][5][3]
// ---------------------------------------------------------------------------
__global__ void __launch_bounds__(256) coords_kernel(const float* __restrict__ X,
                                                     float* __restrict__ coords5) {
    int r = blockIdx.x * blockDim.x + threadIdx.x;
    if (r >= LRES) return;
    const float* xr = X + r * 37 * 3;
    float N[3], Ca[3], C[3], O[3];
    for (int c = 0; c < 3; ++c) {
        N[c]  = xr[0 * 3 + c];
        Ca[c] = xr[1 * 3 + c];
        C[c]  = xr[2 * 3 + c];
        O[c]  = xr[4 * 3 + c];
    }
    float b[3], cc[3];
    for (int c = 0; c < 3; ++c) { b[c] = Ca[c] - N[c]; cc[c] = C[c] - Ca[c]; }
    float a0 = b[1]*cc[2] - b[2]*cc[1];
    float a1 = b[2]*cc[0] - b[0]*cc[2];
    float a2 = b[0]*cc[1] - b[1]*cc[0];
    float Cb[3];
    Cb[0] = -0.58273431f*a0 + 0.56802827f*b[0] - 0.54067466f*cc[0] + Ca[0];
    Cb[1] = -0.58273431f*a1 + 0.56802827f*b[1] - 0.54067466f*cc[1] + Ca[1];
    Cb[2] = -0.58273431f*a2 + 0.56802827f*b[2] - 0.54067466f*cc[2] + Ca[2];
    float* o = coords5 + r * 15;
    for (int c = 0; c < 3; ++c) {
        o[0*3+c] = N[c]; o[1*3+c] = Ca[c]; o[2*3+c] = C[c]; o[3*3+c] = O[c]; o[4*3+c] = Cb[c];
    }
}

// ---------------------------------------------------------------------------
// Kernel 2: f32 -> bf16 weight conversion; WbE is zero-padded 128 x 512
// ---------------------------------------------------------------------------
__global__ void __launch_bounds__(256) convert_weights(const float* __restrict__ We,
                                                       const float* __restrict__ Ws,
                                                       unsigned short* __restrict__ WbE,
                                                       unsigned short* __restrict__ WbS) {
    int i = blockIdx.x * blockDim.x + threadIdx.x;
    if (i < EF * KEP) {
        int o = i / KEP, k = i % KEP;
        WbE[i] = (k < KE) ? f32_to_bf16(We[o * KE + k]) : (unsigned short)0;
    }
    if (i < EF * KS)  WbS[i] = f32_to_bf16(Ws[i]);
}

// ---------------------------------------------------------------------------
// Kernel 3: masked pairwise Ca distances + stable top-48 (matches lax.top_k)
// one block per center residue
// ---------------------------------------------------------------------------
__global__ void __launch_bounds__(256) topk_kernel(const float* __restrict__ coords5,
                                                   const float* __restrict__ mask,
                                                   int*   __restrict__ eidx,
                                                   float* __restrict__ dnb,
                                                   float* __restrict__ outEidx,
                                                   float* __restrict__ outEidxSub) {
    __shared__ float dist[LRES];
    __shared__ unsigned long long red[256];
    __shared__ float fred[256];

    int i = blockIdx.x;
    int tid = threadIdx.x;
    float cx = coords5[i*15 + 3], cy = coords5[i*15 + 4], cz = coords5[i*15 + 5];
    float mi = mask[i];

    float lmax = -1e30f;
    for (int t = 0; t < LRES / 256; ++t) {
        int j = tid + t * 256;
        float dx = cx - coords5[j*15 + 3];
        float dy = cy - coords5[j*15 + 4];
        float dz = cz - coords5[j*15 + 5];
        float m2 = mi * mask[j];
        float D = m2 * sqrtf(dx*dx + dy*dy + dz*dz + 1e-6f);
        dist[j] = D;
        lmax = fmaxf(lmax, D);
    }
    fred[tid] = lmax;
    __syncthreads();
    for (int s = 128; s > 0; s >>= 1) {
        if (tid < s) fred[tid] = fmaxf(fred[tid], fred[tid + s]);
        __syncthreads();
    }
    float Dmax = fred[0];
    __syncthreads();
    for (int t = 0; t < LRES / 256; ++t) {
        int j = tid + t * 256;
        float m2 = mi * mask[j];
        dist[j] += (1.0f - m2) * Dmax;          // D_adj
    }
    __syncthreads();

    for (int it = 0; it < TOPK; ++it) {
        unsigned long long best = ~0ull;
        for (int t = 0; t < LRES / 256; ++t) {
            int j = tid + t * 256;
            unsigned long long key =
                (((unsigned long long)__float_as_uint(dist[j])) << 32) | (unsigned)j;
            best = best < key ? best : key;
        }
        red[tid] = best;
        __syncthreads();
        for (int s = 128; s > 0; s >>= 1) {
            if (tid < s) red[tid] = red[tid] < red[tid + s] ? red[tid] : red[tid + s];
            __syncthreads();
        }
        if (tid == 0) {
            unsigned long long k = red[0];
            int widx = (int)(k & 0xffffffffu);
            float dv = __uint_as_float((unsigned)(k >> 32));
            int e = i * TOPK + it;
            eidx[e] = widx;
            dnb[e]  = dv;
            outEidx[e]    = (float)widx;
            outEidxSub[e] = (float)widx;        // Ks = min(100,48) = 48
            dist[widx] = __int_as_float(0x7f800000); // +inf: remove from pool
        }
        __syncthreads();
    }
}

// ---------------------------------------------------------------------------
// Kernel 4: edge features (416, padded 512) fused with GEMM x W_e^T + LayerNorm
// one WG = 16 edges (one center residue l), 8 waves x 16 out-channels
// phases of 4 K-chunks: 4 back-to-back WMMAs per wave between barriers
// ---------------------------------------------------------------------------
__global__ void __launch_bounds__(256) edge_gemm_kernel(
    const float* __restrict__ coords5, const int* __restrict__ eidx,
    const float* __restrict__ dnb, const int* __restrict__ residx,
    const int* __restrict__ chains, const float* __restrict__ Wpos,
    const float* __restrict__ bpos, const unsigned short* __restrict__ WbE,
    const float* __restrict__ g_e, const float* __restrict__ be_e,
    float* __restrict__ outE) {

    __shared__ int   sJ[16];
    __shared__ int   sDI[16];
    __shared__ float sDn[16];
    __shared__ float sAnch[15];
    __shared__ float sNC[16][15];
    __shared__ float sPD[16][24];
    __shared__ unsigned short sA[16][128];   // one 4-chunk phase of A
    __shared__ float sC[16][128];

    int tid = threadIdx.x;
    int tIdx = blockIdx.x;
    int l = tIdx / 3;
    int k0 = (tIdx % 3) * 16;

    if (tid < 15) sAnch[tid] = coords5[l*15 + tid];
    if (tid < 16) {
        int j = eidx[l*TOPK + k0 + tid];
        sJ[tid]  = j;
        sDn[tid] = dnb[l*TOPK + k0 + tid];
        int off = residx[l] - residx[j];
        int dcl = off + 32; dcl = dcl < 0 ? 0 : (dcl > 64 ? 64 : dcl);
        sDI[tid] = (chains[l] == chains[j]) ? dcl : 65;
    }
    __syncthreads();
    for (int idx = tid; idx < 16*15; idx += 256) {
        int m = idx / 15, c = idx % 15;
        sNC[m][c] = coords5[sJ[m]*15 + c];
    }
    __syncthreads();
    for (int idx = tid; idx < 16*24; idx += 256) {
        int m = idx / 24, p = idx % 24;
        int a = PAIR_A[p], b = PAIR_B[p];
        float dx = sAnch[a*3+0] - sNC[m][b*3+0];
        float dy = sAnch[a*3+1] - sNC[m][b*3+1];
        float dz = sAnch[a*3+2] - sNC[m][b*3+2];
        sPD[m][p] = sqrtf(dx*dx + dy*dy + dz*dz + 1e-6f);
    }
    __syncthreads();

    int wave = tid >> 5, lane = tid & 31;
    int nbase = wave * 16;
    int n = lane & 15;
    int mr = lane & 15;
    int khalf = (lane >> 4) * 8;
    const unsigned short* wrow = WbE + (nbase + n) * KEP;
    v8f acc = {};

    for (int ph = 0; ph < KEP / 128; ++ph) {
        // build 16 x 128 A sub-tile (8 features per thread)
        for (int idx = tid; idx < 16 * 128; idx += 256) {
            int m = idx >> 7, col = idx & 127;
            int f = ph * 128 + col;
            float v = 0.0f;
            if (f < 16) {
                v = Wpos[f * 66 + sDI[m]] + bpos[f];
            } else if (f < 32) {
                int r = f - 16;
                float mu = 2.0f + (20.0f / 15.0f) * r;
                float z = (sDn[m] - mu) * (1.0f / 1.25f);
                v = __expf(-z * z);
            } else if (f < KE) {
                int p = (f - 32) >> 4, r = (f - 32) & 15;
                float mu = 2.0f + (20.0f / 15.0f) * r;
                float z = (sPD[m][p] - mu) * (1.0f / 1.25f);
                v = __expf(-z * z);
            }
            sA[m][col] = f32_to_bf16(v);
        }
        __syncthreads();

        // prefetch next phase's weight block into cache
        if (ph + 1 < KEP / 128)
            __builtin_prefetch(wrow + (ph + 1) * 128, 0, 1);

        #pragma unroll
        for (int c = 0; c < 4; ++c) {
            // A fragment (16-bit A 16x32 layout)
            unsigned short ta[16];
            #pragma unroll
            for (int i = 0; i < 16; ++i) {
                int k = ((i < 8) ? 0 : 16) + khalf + (i & 7);
                ta[i] = sA[mr][c * 32 + k];
            }
            v16bf af; __builtin_memcpy(&af, ta, sizeof(af));

            // B fragment from bf16 weights (row = output channel)
            unsigned tb[8];
            const unsigned short* wp = wrow + ph * 128 + c * 32;
            #pragma unroll
            for (int ip = 0; ip < 8; ++ip) {
                int kk = ((ip < 4) ? 0 : 16) + khalf + (ip & 3) * 2;
                tb[ip] = *(const unsigned*)(wp + kk);
            }
            v16bf bfv; __builtin_memcpy(&bfv, tb, sizeof(bfv));

            acc = __builtin_amdgcn_wmma_f32_16x16x32_bf16(
                false, af, false, bfv, (short)0, acc, false, false);
        }
        __syncthreads();
    }

    {
        int mrow = (lane >> 4) * 8;
        for (int v = 0; v < 8; ++v)
            sC[v + mrow][nbase + n] = acc[v];
    }
    __syncthreads();

    // fused LayerNorm: wave w handles rows w and w+8
    for (int rr = 0; rr < 2; ++rr) {
        int m = wave + rr * 8;
        float x[4];
        for (int q = 0; q < 4; ++q) x[q] = sC[m][lane * 4 + q];
        float s = x[0] + x[1] + x[2] + x[3];
        for (int off = 16; off > 0; off >>= 1) s += __shfl_xor(s, off, 32);
        float mu = s * (1.0f / 128.0f);
        float vs = 0.f;
        for (int q = 0; q < 4; ++q) { float d = x[q] - mu; vs += d * d; }
        for (int off = 16; off > 0; off >>= 1) vs += __shfl_xor(vs, off, 32);
        float inv = rsqrtf(vs * (1.0f / 128.0f) + 1e-5f);
        int erow = (l * TOPK + k0 + m) * EF;
        for (int q = 0; q < 4; ++q) {
            int nn = lane * 4 + q;
            outE[erow + nn] = g_e[nn] * (x[q] - mu) * inv + be_e[nn];
        }
    }
}

// ---------------------------------------------------------------------------
// Kernel 5: side-chain features (1280) fused with GEMM x W_s^T + LayerNorm
// phases of 4 K-chunks: 4 back-to-back WMMAs per wave between barriers
// ---------------------------------------------------------------------------
__global__ void __launch_bounds__(256) sc_gemm_kernel(
    const float* __restrict__ coords5, const float* __restrict__ X,
    const float* __restrict__ atom_mask, const int* __restrict__ eidx,
    const unsigned short* __restrict__ WbS, const float* __restrict__ g_s,
    const float* __restrict__ b_s, float* __restrict__ outEs) {

    __shared__ float sAnch[15];
    __shared__ int   sJ[16];
    __shared__ float sSnb[16][32][3];
    __shared__ float sAm[16][32];
    __shared__ float sDist[16][160];     // [m][a*32+s]
    __shared__ unsigned short sA[16][128];
    __shared__ float sC[16][128];

    int tid = threadIdx.x;
    int tIdx = blockIdx.x;
    int l = tIdx / 3;
    int k0 = (tIdx % 3) * 16;

    if (tid < 15) sAnch[tid] = coords5[l*15 + tid];
    if (tid < 16) sJ[tid] = eidx[l*TOPK + k0 + tid];
    __syncthreads();

    for (int idx = tid; idx < 16*32*3; idx += 256) {
        int m = idx / 96, rem = idx % 96, s = rem / 3, c = rem % 3;
        sSnb[m][s][c] = X[sJ[m] * 111 + (5 + s) * 3 + c];
    }
    for (int idx = tid; idx < 16*32; idx += 256) {
        int m = idx >> 5, s = idx & 31;
        sAm[m][s] = atom_mask[sJ[m] * 37 + 5 + s];
    }
    __syncthreads();
    for (int idx = tid; idx < 16*160; idx += 256) {
        int m = idx / 160, rem = idx % 160, a = rem / 32, s = rem & 31;
        float dx = sAnch[a*3+0] - sSnb[m][s][0];
        float dy = sAnch[a*3+1] - sSnb[m][s][1];
        float dz = sAnch[a*3+2] - sSnb[m][s][2];
        sDist[m][rem] = sqrtf(dx*dx + dy*dy + dz*dz + 1e-6f);
    }
    __syncthreads();

    int wave = tid >> 5, lane = tid & 31;
    int nbase = wave * 16;
    int n = lane & 15;
    int mr = lane & 15;
    int khalf = (lane >> 4) * 8;
    const unsigned short* wrow = WbS + (nbase + n) * KS;
    v8f acc = {};

    for (int ph = 0; ph < KS / 128; ++ph) {
        for (int idx = tid; idx < 16 * 128; idx += 256) {
            int m = idx >> 7, col = idx & 127;
            int f = ph * 128 + col;              // f = ((a*32)+s)*8 + r
            int a = f >> 8, s = (f >> 3) & 31, r = f & 7;
            float mu = 2.0f + (20.0f / 7.0f) * r;
            float z = (sDist[m][a * 32 + s] - mu) * (1.0f / 2.5f);
            sA[m][col] = f32_to_bf16(__expf(-z * z) * sAm[m][s]);
        }
        __syncthreads();

        if (ph + 1 < KS / 128)
            __builtin_prefetch(wrow + (ph + 1) * 128, 0, 1);

        #pragma unroll
        for (int c = 0; c < 4; ++c) {
            unsigned short ta[16];
            #pragma unroll
            for (int i = 0; i < 16; ++i) {
                int k = ((i < 8) ? 0 : 16) + khalf + (i & 7);
                ta[i] = sA[mr][c * 32 + k];
            }
            v16bf af; __builtin_memcpy(&af, ta, sizeof(af));

            unsigned tb[8];
            const unsigned short* wp = wrow + ph * 128 + c * 32;
            #pragma unroll
            for (int ip = 0; ip < 8; ++ip) {
                int kk = ((ip < 4) ? 0 : 16) + khalf + (ip & 3) * 2;
                tb[ip] = *(const unsigned*)(wp + kk);
            }
            v16bf bfv; __builtin_memcpy(&bfv, tb, sizeof(bfv));

            acc = __builtin_amdgcn_wmma_f32_16x16x32_bf16(
                false, af, false, bfv, (short)0, acc, false, false);
        }
        __syncthreads();
    }

    {
        int mrow = (lane >> 4) * 8;
        for (int v = 0; v < 8; ++v)
            sC[v + mrow][nbase + n] = acc[v];
    }
    __syncthreads();

    for (int rr = 0; rr < 2; ++rr) {
        int m = wave + rr * 8;
        float x[4];
        for (int q = 0; q < 4; ++q) x[q] = sC[m][lane * 4 + q];
        float s = x[0] + x[1] + x[2] + x[3];
        for (int off = 16; off > 0; off >>= 1) s += __shfl_xor(s, off, 32);
        float mu = s * (1.0f / 128.0f);
        float vs = 0.f;
        for (int q = 0; q < 4; ++q) { float d = x[q] - mu; vs += d * d; }
        for (int off = 16; off > 0; off >>= 1) vs += __shfl_xor(vs, off, 32);
        float inv = rsqrtf(vs * (1.0f / 128.0f) + 1e-5f);
        int erow = (l * TOPK + k0 + m) * EF;
        for (int q = 0; q < 4; ++q) {
            int nn = lane * 4 + q;
            outEs[erow + nn] = g_s[nn] * (x[q] - mu) * inv + b_s[nn];
        }
    }
}

// ---------------------------------------------------------------------------
extern "C" void kernel_launch(void* const* d_in, const int* in_sizes, int n_in,
                              void* d_out, int out_size, void* d_ws, size_t ws_size,
                              hipStream_t stream) {
    const float* X         = (const float*)d_in[0];
    const float* mask      = (const float*)d_in[2];
    const float* atom_mask = (const float*)d_in[3];
    const int*   residx    = (const int*)d_in[4];
    const int*   chains    = (const int*)d_in[6];
    const float* Wpos      = (const float*)d_in[7];
    const float* bpos      = (const float*)d_in[8];
    const float* We        = (const float*)d_in[9];
    const float* ge        = (const float*)d_in[10];
    const float* bee       = (const float*)d_in[11];
    const float* Ws        = (const float*)d_in[12];
    const float* gs        = (const float*)d_in[13];
    const float* bs        = (const float*)d_in[14];

    // workspace layout (< 1 MB total)
    float* coords5          = (float*)d_ws;                   // 1024*15 f32
    int*   eidx             = (int*)(coords5 + LRES * 15);    // 49152 i32
    float* dnb              = (float*)(eidx + NEDGE);         // 49152 f32
    unsigned short* WbE     = (unsigned short*)(dnb + NEDGE); // 128*512 bf16 (padded)
    unsigned short* WbS     = WbE + EF * KEP;                 // 128*1280 bf16

    float* out        = (float*)d_out;
    float* outE       = out;                                  // 1024*48*128
    float* outEidx    = outE + (long long)NEDGE * EF;         // 49152
    float* outEs      = outEidx + NEDGE;                      // 1024*48*128
    float* outEidxSub = outEs + (long long)NEDGE * EF;        // 49152

    coords_kernel<<<(LRES + 255) / 256, 256, 0, stream>>>(X, coords5);
    convert_weights<<<(EF * KS + 255) / 256, 256, 0, stream>>>(We, Ws, WbE, WbS);
    topk_kernel<<<LRES, 256, 0, stream>>>(coords5, mask, eidx, dnb, outEidx, outEidxSub);
    edge_gemm_kernel<<<NEDGE / 16, 256, 0, stream>>>(coords5, eidx, dnb, residx, chains,
                                                     Wpos, bpos, WbE, ge, bee, outE);
    sc_gemm_kernel<<<NEDGE / 16, 256, 0, stream>>>(coords5, X, atom_mask, eidx,
                                                   WbS, gs, bs, outEs);
}
